// SoftPoolFeat_2422361555207
// MI455X (gfx1250) — matrix operand order repaired and checked by reading, not compiled, not run on hardware
//
#include <hip/hip_runtime.h>
#include <hip/hip_bf16.h>
#include <stddef.h>

// ---------------------------------------------------------------------------
// SoftPoolFeat for MI455X (gfx1250, wave32, WMMA).
//
//   0. init vmax, pack conv weights into lane-swizzled f16 WMMA A-tiles
//   1. stn_chain : x(15x2048) -> 64 -> 128 -> 1024 (WMMA, fused BN+ReLU),
//                  fused global max over N via shfl_xor + atomicMax(bits)
//   2. stn_fc    : 1024 -> 512 -> 256 -> 225 (+eye) -> trans
//   3. transform : xt[j,n] = sum_k x[k,n] * trans[k,j]
//   4. feat_chain: xt -> 64 -> 128 -> 256 (WMMA, last layer BN only) -> f
//   5. argsort   : stable bitonic sort (desc key, asc idx), 64 rows in LDS
//   6. gather    : per (c,r): sp_cube row + 8 cabin maxes + sp_window row
//   7. part      : part rows (xt gathered) + sp_idx broadcast
//
// LDS activations live in B-operand-native f16 layout:
//   act16[(k>>4)*256 + n*16 + (k&15)]
// so every WMMA B operand is ONE aligned 32-byte LDS read per lane
// (no divergent guards, no per-element cvt chains).
// ---------------------------------------------------------------------------

typedef __attribute__((ext_vector_type(16))) _Float16 v16h;
typedef __attribute__((ext_vector_type(8)))  float    v8f;

union HalfPack { _Float16 h[2]; unsigned u; };

#define BN_RSQ 0.9999950000374997f  /* 1/sqrt(1+1e-5) */

#define N_PTS   2048
#define KDIM    15
#define REGIONS 64
#define NCABIN  8
#define PC      (N_PTS / NCABIN)    /* 256 */

// ---------------- weight packing: f32 (Cout x Cin) -> swizzled f16 A tiles ---

__global__ __launch_bounds__(32) void pack_w_kernel(const float* __restrict__ W,
                                                    int Cin, int numK,
                                                    _Float16* __restrict__ out) {
  int tile = blockIdx.x;            // tile = mt*numK + kt
  int lane = threadIdx.x;           // 0..31
  int mt = tile / numK, kt = tile % numK;
  int m = mt * 16 + (lane & 15);
  int h = lane >> 4;
  _Float16* dst = out + (size_t)tile * 512 + lane * 16;
#pragma unroll
  for (int j = 0; j < 16; ++j) {
    int k = kt * 32 + (h << 4) + j;
    float v = (k < Cin) ? W[(size_t)m * Cin + k] : 0.0f;
    dst[j] = (_Float16)v;
  }
}

__global__ void init_vmax_kernel(unsigned* __restrict__ vmaxBits) {
  int t = blockIdx.x * 256 + threadIdx.x;
  if (t < 1024) vmaxBits[t] = 0u;   // post-ReLU max >= 0 -> 0.0f bits
}

// ---------------- x-tile load into B-native f16 layout -----------------------

__device__ inline void load_x_tile(const float* __restrict__ src, int n0,
                                   _Float16* act16, int lane) {
  // zero sub-tile 1 (k = 16..31): 256 halves = 128 dwords
  unsigned* z = (unsigned*)(act16 + 256);
  for (int t = lane; t < 128; t += 32) z[t] = 0u;
  // fill sub-tile 0: act16[n*16 + k] = src[k][n0+n] for k<15, 0 for k==15
  for (int t = lane; t < 256; t += 32) {
    int n = t >> 4, k = t & 15;
    float v = src[(size_t)(k < KDIM ? k : 0) * N_PTS + n0 + n];  // safe addr
    if (k >= KDIM) v = 0.0f;                                     // cndmask
    act16[n * 16 + k] = (_Float16)v;
  }
}

// ---------------- conv layers (one 16-point tile per wave) -------------------

// y = relu((Wx+b)*s + be), output back to LDS in B-native f16 layout.
template <bool RELU>
__device__ inline void conv_layer_lds(const _Float16* __restrict__ pack,
                                      int Cout, int numK,
                                      const float* __restrict__ g,
                                      const float* __restrict__ be,
                                      const float* __restrict__ bc,
                                      const _Float16* actIn16,
                                      _Float16* actOut16, int lane) {
  const v16h* aP = (const v16h*)pack;
  const v16h* bP = (const v16h*)actIn16;
  int h = lane >> 4, n = lane & 15;
  int numM = Cout >> 4;
  for (int mt = 0; mt < numM; ++mt) {
    v8f acc = {0.f, 0.f, 0.f, 0.f, 0.f, 0.f, 0.f, 0.f};
    for (int kt = 0; kt < numK; ++kt) {
      v16h a = aP[(size_t)(mt * numK + kt) * 32 + lane];
      v16h b = bP[(kt * 2 + h) * 16 + n];     // one 32B aligned LDS read
      acc = __builtin_amdgcn_wmma_f32_16x16x32_f16(false, a, false, b,
                                                   (short)0, acc, false, false);
    }
    // D element (m = mt*16 + h*8 + r, n) -> half offset mt*256 + n*16 + h*8 + r
    unsigned* outw = (unsigned*)(actOut16 + mt * 256 + n * 16 + h * 8);
#pragma unroll
    for (int rr = 0; rr < 4; ++rr) {
      int m0 = mt * 16 + h * 8 + rr * 2;
      float v0 = (acc[2 * rr]     + bc[m0])     * (g[m0]     * BN_RSQ) + be[m0];
      float v1 = (acc[2 * rr + 1] + bc[m0 + 1]) * (g[m0 + 1] * BN_RSQ) + be[m0 + 1];
      if (RELU) { v0 = fmaxf(v0, 0.0f); v1 = fmaxf(v1, 0.0f); }
      HalfPack p;
      p.h[0] = (_Float16)v0;
      p.h[1] = (_Float16)v1;
      outw[rr] = p.u;                          // packed ds_store_b32
    }
  }
}

// Conv layer writing f32 straight to global (Cout x 2048 row-major).
template <bool RELU>
__device__ inline void conv_layer_gmem(const _Float16* __restrict__ pack,
                                       int Cout, int numK,
                                       const float* __restrict__ g,
                                       const float* __restrict__ be,
                                       const float* __restrict__ bc,
                                       const _Float16* actIn16,
                                       float* __restrict__ gout, int n0,
                                       int lane) {
  const v16h* aP = (const v16h*)pack;
  const v16h* bP = (const v16h*)actIn16;
  int h = lane >> 4, n = lane & 15;
  int numM = Cout >> 4;
  for (int mt = 0; mt < numM; ++mt) {
    v8f acc = {0.f, 0.f, 0.f, 0.f, 0.f, 0.f, 0.f, 0.f};
    for (int kt = 0; kt < numK; ++kt) {
      v16h a = aP[(size_t)(mt * numK + kt) * 32 + lane];
      v16h b = bP[(kt * 2 + h) * 16 + n];
      acc = __builtin_amdgcn_wmma_f32_16x16x32_f16(false, a, false, b,
                                                   (short)0, acc, false, false);
    }
#pragma unroll
    for (int r = 0; r < 8; ++r) {
      int m = mt * 16 + h * 8 + r;
      float v = (acc[r] + bc[m]) * (g[m] * BN_RSQ) + be[m];
      if (RELU) v = fmaxf(v, 0.0f);
      gout[(size_t)m * N_PTS + n0 + n] = v;
    }
  }
}

// Final STN layer (128 -> 1024) with fused max-over-N (16 cols in this tile),
// reduced across lanes of each half-wave, then bit-atomicMax into vmax[1024].
__device__ inline void conv_layer_max(const _Float16* __restrict__ pack,
                                      const float* __restrict__ g,
                                      const float* __restrict__ be,
                                      const float* __restrict__ bc,
                                      const _Float16* actIn16,
                                      unsigned* __restrict__ vmaxBits,
                                      int lane) {
  const v16h* aP = (const v16h*)pack;
  const v16h* bP = (const v16h*)actIn16;
  int h = lane >> 4, n = lane & 15;
  for (int mt = 0; mt < 64; ++mt) {
    v8f acc = {0.f, 0.f, 0.f, 0.f, 0.f, 0.f, 0.f, 0.f};
    for (int kt = 0; kt < 4; ++kt) {
      v16h a = aP[(size_t)(mt * 4 + kt) * 32 + lane];
      v16h b = bP[(kt * 2 + h) * 16 + n];
      acc = __builtin_amdgcn_wmma_f32_16x16x32_f16(false, a, false, b,
                                                   (short)0, acc, false, false);
    }
#pragma unroll
    for (int r = 0; r < 8; ++r) {
      int m = mt * 16 + h * 8 + r;
      float v = (acc[r] + bc[m]) * (g[m] * BN_RSQ) + be[m];
      v = fmaxf(v, 0.0f);
      // max over the 16 columns held by this half-wave (xor < 16 stays in half)
      v = fmaxf(v, __shfl_xor(v, 1, 32));
      v = fmaxf(v, __shfl_xor(v, 2, 32));
      v = fmaxf(v, __shfl_xor(v, 4, 32));
      v = fmaxf(v, __shfl_xor(v, 8, 32));
      if ((lane & 15) == 0)
        atomicMax(vmaxBits + m, __float_as_uint(v));  // valid: v >= 0
    }
  }
}

// ---------------- kernel 1: STN conv chain + global max ----------------------

#define NW 4  // waves per block; each wave owns one 16-point tile

__global__ __launch_bounds__(NW * 32) void stn_chain_kernel(
    const float* __restrict__ x, const _Float16* __restrict__ pS1,
    const _Float16* __restrict__ pS2, const _Float16* __restrict__ pS3,
    const float* __restrict__ sb1, const float* __restrict__ sg1,
    const float* __restrict__ sbe1, const float* __restrict__ sb2,
    const float* __restrict__ sg2, const float* __restrict__ sbe2,
    const float* __restrict__ sb3, const float* __restrict__ sg3,
    const float* __restrict__ sbe3, unsigned* __restrict__ vmaxBits) {
  __shared__ __align__(32) _Float16 buf[NW][2][2048];  // 32 KB ping-pong tiles
  int lane = threadIdx.x & 31;
  int w = threadIdx.x >> 5;
  int tile = blockIdx.x * NW + w;  // 0..127
  int n0 = tile * 16;
  _Float16* bA = buf[w][0];
  _Float16* bB = buf[w][1];
  load_x_tile(x, n0, bA, lane);
  conv_layer_lds<true>(pS1, 64, 1, sg1, sbe1, sb1, bA, bB, lane);
  conv_layer_lds<true>(pS2, 128, 2, sg2, sbe2, sb2, bB, bA, lane);
  conv_layer_max(pS3, sg3, sbe3, sb3, bA, vmaxBits, lane);
}

// ---------------- kernel 2: STN fully-connected head -------------------------

__global__ __launch_bounds__(512) void stn_fc_kernel(
    const unsigned* __restrict__ vmaxBits, const float* __restrict__ fc1w,
    const float* __restrict__ fc1b, const float* __restrict__ g4,
    const float* __restrict__ be4, const float* __restrict__ fc2w,
    const float* __restrict__ fc2b, const float* __restrict__ g5,
    const float* __restrict__ be5, const float* __restrict__ fc3w,
    const float* __restrict__ fc3b, float* __restrict__ transWs,
    float* __restrict__ transOut) {
  __shared__ float v[1024];
  __shared__ float h1[512];
  __shared__ float h2[256];
  int t = threadIdx.x;
  for (int i = t; i < 1024; i += 512) v[i] = __uint_as_float(vmaxBits[i]);
  __syncthreads();
  if (t < 512) {
    float a = fc1b[t];
    for (int i = 0; i < 1024; ++i) a += v[i] * fc1w[(size_t)t * 1024 + i];
    a = a * (g4[t] * BN_RSQ) + be4[t];
    h1[t] = fmaxf(a, 0.0f);
  }
  __syncthreads();
  if (t < 256) {
    float a = fc2b[t];
    for (int i = 0; i < 512; ++i) a += h1[i] * fc2w[(size_t)t * 512 + i];
    a = a * (g5[t] * BN_RSQ) + be5[t];
    h2[t] = fmaxf(a, 0.0f);
  }
  __syncthreads();
  if (t < KDIM * KDIM) {
    float a = fc3b[t];
    for (int i = 0; i < 256; ++i) a += h2[i] * fc3w[(size_t)t * 256 + i];
    if (t % (KDIM + 1) == 0) a += 1.0f;  // + eye(15)
    transWs[t] = a;
    transOut[t] = a;
  }
}

// ---------------- kernel 3: xt = trans^T applied to x ------------------------

__global__ __launch_bounds__(256) void transform_kernel(
    const float* __restrict__ x, const float* __restrict__ trans,
    float* __restrict__ xt) {
  __shared__ float tr[KDIM * KDIM];
  if (threadIdx.x < KDIM * KDIM) tr[threadIdx.x] = trans[threadIdx.x];
  __syncthreads();
  int tid = blockIdx.x * 256 + threadIdx.x;
  if (tid < KDIM * N_PTS) {
    int j = tid >> 11, n = tid & (N_PTS - 1);
    float a = 0.0f;
#pragma unroll
    for (int k = 0; k < KDIM; ++k) a += x[(size_t)k * N_PTS + n] * tr[k * KDIM + j];
    xt[(size_t)j * N_PTS + n] = a;
  }
}

// ---------------- kernel 4: feature conv chain -> f (256 x 2048) -------------

__global__ __launch_bounds__(NW * 32) void feat_chain_kernel(
    const float* __restrict__ xt, const _Float16* __restrict__ pW1,
    const _Float16* __restrict__ pW2, const _Float16* __restrict__ pW3,
    const float* __restrict__ b1, const float* __restrict__ g1,
    const float* __restrict__ be1, const float* __restrict__ b2,
    const float* __restrict__ g2, const float* __restrict__ be2,
    const float* __restrict__ b3, const float* __restrict__ g3,
    const float* __restrict__ be3, float* __restrict__ fOut) {
  __shared__ __align__(32) _Float16 buf[NW][2][2048];
  int lane = threadIdx.x & 31;
  int w = threadIdx.x >> 5;
  int tile = blockIdx.x * NW + w;
  int n0 = tile * 16;
  _Float16* bA = buf[w][0];
  _Float16* bB = buf[w][1];
  load_x_tile(xt, n0, bA, lane);
  conv_layer_lds<true>(pW1, 64, 1, g1, be1, b1, bA, bB, lane);
  conv_layer_lds<true>(pW2, 128, 2, g2, be2, b2, bB, bA, lane);
  conv_layer_gmem<false>(pW3, 256, 4, g3, be3, b3, bA, fOut, n0, lane);
}

// ---------------- kernel 5: stable descending argsort per key row ------------

__global__ __launch_bounds__(512) void argsort_kernel(
    const float* __restrict__ f, int* __restrict__ gIdx) {
  __shared__ float sk[N_PTS];
  __shared__ int si[N_PTS];
  int r = blockIdx.x;  // 0..63 (keys = f rows 0..63)
  for (int n = threadIdx.x; n < N_PTS; n += 512) {
    sk[n] = f[(size_t)r * N_PTS + n];
    si[n] = n;
  }
  __syncthreads();
  for (int k = 2; k <= N_PTS; k <<= 1) {
    for (int j = k >> 1; j > 0; j >>= 1) {
      for (int i = threadIdx.x; i < N_PTS; i += 512) {
        int ixj = i ^ j;
        if (ixj > i) {
          bool up = (i & k) == 0;
          float ka = sk[i], kb = sk[ixj];
          int ia = si[i], ib = si[ixj];
          // order: larger key first; equal keys -> smaller index first
          bool b_first = (kb > ka) || (kb == ka && ib < ia);
          bool a_first = (ka > kb) || (ka == kb && ia < ib);
          if (up ? b_first : a_first) {
            sk[i] = kb; sk[ixj] = ka;
            si[i] = ib; si[ixj] = ia;
          }
        }
      }
      __syncthreads();
    }
  }
  for (int n = threadIdx.x; n < N_PTS; n += 512)
    gIdx[(size_t)r * N_PTS + n] = si[n];
}

// ---------------- kernel 6: sp_cube gather + cabins + sp_window --------------

__global__ __launch_bounds__(PC) void gather_kernel(
    const float* __restrict__ f, const int* __restrict__ gIdx,
    float* __restrict__ feature, float* __restrict__ cabins) {
  int c = blockIdx.x;  // 0..255 feature channel
  int r = blockIdx.y;  // 0..63 region
  __shared__ float red[PC];
  const int* idxr = gIdx + (size_t)r * N_PTS;
  const float* fr = f + (size_t)c * N_PTS;
  float* featc = feature + ((size_t)c * REGIONS + r) * N_PTS;          // sp_cube
  float* featw = feature + ((size_t)(256 + c) * REGIONS + r) * N_PTS;  // window
  for (int it = 0; it < NCABIN; ++it) {
    int n = it * PC + threadIdx.x;  // each iteration is exactly one cabin chunk
    __builtin_prefetch(idxr + (((it + 1) & (NCABIN - 1)) * PC + threadIdx.x), 0, 0);
    int i = idxr[n];
    float v = fr[i];
    featc[n] = v;
    red[threadIdx.x] = v;
    __syncthreads();
    for (int s = PC / 2; s > 0; s >>= 1) {
      if (threadIdx.x < (unsigned)s)
        red[threadIdx.x] = fmaxf(red[threadIdx.x], red[threadIdx.x + s]);
      __syncthreads();
    }
    float cm = red[0];
    if (threadIdx.x == 0)
      cabins[((size_t)c * REGIONS + r) * NCABIN + it] = cm;
    featw[n] = cm;
    __syncthreads();
  }
}

// ---------------- kernel 7: part gather + sp_idx broadcast -------------------

__global__ __launch_bounds__(256) void part_kernel(
    const float* __restrict__ xt, const int* __restrict__ gIdx,
    float* __restrict__ feature, float* __restrict__ spidx) {
  int r = blockIdx.x;
  for (int n = threadIdx.x; n < N_PTS; n += 256) {
    int i = gIdx[(size_t)r * N_PTS + n];
    float fi = (float)i;
#pragma unroll
    for (int k = 0; k < KDIM; ++k) {
      feature[((size_t)(512 + k) * REGIONS + r) * N_PTS + n] =
          xt[(size_t)k * N_PTS + i];
      spidx[((size_t)k * REGIONS + r) * N_PTS + n] = fi;
    }
  }
}

// ---------------- host-side orchestration ------------------------------------

extern "C" void kernel_launch(void* const* d_in, const int* in_sizes, int n_in,
                              void* d_out, int out_size, void* d_ws,
                              size_t ws_size, hipStream_t stream) {
  (void)in_sizes; (void)n_in; (void)out_size; (void)ws_size;

  // inputs in setup_inputs() dict order: x, then params in insertion order
  const float* x      = (const float*)d_in[0];
  const float* s_w1   = (const float*)d_in[1];
  const float* s_b1   = (const float*)d_in[2];
  const float* s_w2   = (const float*)d_in[3];
  const float* s_b2   = (const float*)d_in[4];
  const float* s_w3   = (const float*)d_in[5];
  const float* s_b3   = (const float*)d_in[6];
  const float* s_fc1w = (const float*)d_in[7];
  const float* s_fc1b = (const float*)d_in[8];
  const float* s_fc2w = (const float*)d_in[9];
  const float* s_fc2b = (const float*)d_in[10];
  const float* s_fc3w = (const float*)d_in[11];
  const float* s_fc3b = (const float*)d_in[12];
  const float* s_g1   = (const float*)d_in[13];
  const float* s_be1  = (const float*)d_in[14];
  const float* s_g2   = (const float*)d_in[15];
  const float* s_be2  = (const float*)d_in[16];
  const float* s_g3   = (const float*)d_in[17];
  const float* s_be3  = (const float*)d_in[18];
  const float* s_g4   = (const float*)d_in[19];
  const float* s_be4  = (const float*)d_in[20];
  const float* s_g5   = (const float*)d_in[21];
  const float* s_be5  = (const float*)d_in[22];
  const float* w1     = (const float*)d_in[23];
  const float* b1     = (const float*)d_in[24];
  const float* w2     = (const float*)d_in[25];
  const float* b2     = (const float*)d_in[26];
  const float* w3     = (const float*)d_in[27];
  const float* b3     = (const float*)d_in[28];
  const float* g1     = (const float*)d_in[29];
  const float* be1    = (const float*)d_in[30];
  const float* g2     = (const float*)d_in[31];
  const float* be2    = (const float*)d_in[32];
  const float* g3     = (const float*)d_in[33];
  const float* be3    = (const float*)d_in[34];

  // output regions (floats, return order: feature, cabins, sp_idx, trans)
  float* out = (float*)d_out;
  const size_t CAB_OFF   = (size_t)527 * REGIONS * N_PTS;          // 69,074,944
  const size_t SPIDX_OFF = CAB_OFF + (size_t)256 * REGIONS * NCABIN;
  const size_t TRANS_OFF = SPIDX_OFF + (size_t)KDIM * REGIONS * N_PTS;
  float* feature  = out;
  float* cabins   = out + CAB_OFF;
  float* spidx    = out + SPIDX_OFF;
  float* transOut = out + TRANS_OFF;

  // workspace layout (bytes): packed f16 weights, then f32/int scratch
  char* ws = (char*)d_ws;
  _Float16* pS1 = (_Float16*)(ws + 0);        //   4 tiles *1KB =   4 KB
  _Float16* pS2 = (_Float16*)(ws + 4096);     //  16 tiles      =  16 KB
  _Float16* pS3 = (_Float16*)(ws + 20480);    // 256 tiles      = 256 KB
  _Float16* pW1 = (_Float16*)(ws + 282624);   //   4 tiles      =   4 KB
  _Float16* pW2 = (_Float16*)(ws + 286720);   //  16 tiles      =  16 KB
  _Float16* pW3 = (_Float16*)(ws + 303104);   //  64 tiles      =  64 KB
  float* wsF      = (float*)(ws + 368640);
  unsigned* vmax  = (unsigned*)wsF;           // 1024
  float* transWs  = wsF + 1024;               // 225 (padded to 256)
  float* xt       = wsF + 1280;               // 15*2048
  float* fbuf     = wsF + 32000;              // 256*2048
  int* gIdx       = (int*)(wsF + 556288);     // 64*2048

  // 0. init + pack weights into WMMA A-operand layout
  init_vmax_kernel<<<4, 256, 0, stream>>>(vmax);
  pack_w_kernel<<<4,   32, 0, stream>>>(s_w1, 15,  1, pS1);
  pack_w_kernel<<<16,  32, 0, stream>>>(s_w2, 64,  2, pS2);
  pack_w_kernel<<<256, 32, 0, stream>>>(s_w3, 128, 4, pS3);
  pack_w_kernel<<<4,   32, 0, stream>>>(w1,   15,  1, pW1);
  pack_w_kernel<<<16,  32, 0, stream>>>(w2,   64,  2, pW2);
  pack_w_kernel<<<64,  32, 0, stream>>>(w3,   128, 4, pW3);

  // 1. STN conv chain + fused global max (128 tiles / NW waves per block)
  stn_chain_kernel<<<128 / NW, NW * 32, 0, stream>>>(
      x, pS1, pS2, pS3, s_b1, s_g1, s_be1, s_b2, s_g2, s_be2, s_b3, s_g3,
      s_be3, vmax);

  // 2. STN FC head -> trans (ws copy + output copy)
  stn_fc_kernel<<<1, 512, 0, stream>>>(vmax, s_fc1w, s_fc1b, s_g4, s_be4,
                                       s_fc2w, s_fc2b, s_g5, s_be5, s_fc3w,
                                       s_fc3b, transWs, transOut);

  // 3. xt = einsum('kn,kj->jn', x, trans)
  transform_kernel<<<(KDIM * N_PTS + 255) / 256, 256, 0, stream>>>(x, transWs,
                                                                   xt);

  // 4. feature conv chain -> f (256 x 2048)
  feat_chain_kernel<<<128 / NW, NW * 32, 0, stream>>>(
      xt, pW1, pW2, pW3, b1, g1, be1, b2, g2, be2, b3, g3, be3, fbuf);

  // 5. stable descending argsort of keys = f[0:64,:]
  argsort_kernel<<<REGIONS, 512, 0, stream>>>(fbuf, gIdx);

  // 6. sp_cube gather + cabins + sp_window (one block per (channel, region))
  gather_kernel<<<dim3(256, REGIONS), PC, 0, stream>>>(fbuf, gIdx, feature,
                                                       cabins);

  // 7. part gather + sp_idx broadcast
  part_kernel<<<REGIONS, 256, 0, stream>>>(xt, gIdx, feature, spidx);
}
